// ACF_model_69337952026709
// MI455X (gfx1250) — matrix operand order repaired and checked by reading, not compiled, not run on hardware
//
#include <hip/hip_runtime.h>
#include <hip/hip_bf16.h>

typedef __attribute__((ext_vector_type(16))) _Float16 v16h;
typedef __attribute__((ext_vector_type(8)))  _Float16 v8h;
typedef __attribute__((ext_vector_type(4)))  _Float16 v4h;
typedef __attribute__((ext_vector_type(8)))  float    v8f;

#define B_   128
#define P_   50
#define F_   200
#define L_   49
#define C_   128
#define K_   64
#define AST  136   // padded LDS row stride (halves): 272B -> +4 banks per row
#define MPAD 64    // A rows padded to 64 (rows 49..63 zeroed)

// ---------------------------------------------------------------------------
// Kernel 1: per-user projections (gamma_u @ Wc0u + bc0, gamma_u @ Wi0u + bi0)
// and the three embedding outputs.
// ---------------------------------------------------------------------------
__global__ __launch_bounds__(128) void user_proj(
    const int* __restrict__ user, const int* __restrict__ item,
    const float* __restrict__ Gu, const float* __restrict__ Gi,
    const float* __restrict__ Pi,
    const float* __restrict__ Wc0u, const float* __restrict__ bc0,
    const float* __restrict__ Wi0u, const float* __restrict__ bi0,
    float* __restrict__ suC, float* __restrict__ suI,
    float* __restrict__ out)
{
    const int b = blockIdx.x;
    const int tid = threadIdx.x;
    __shared__ float sU[F_];
    const int u  = user[b];
    const int it = item[b];
    for (int f = tid; f < F_; f += 128) sU[f] = Gu[(size_t)u * F_ + f];
    __syncthreads();
    if (tid < K_) {
        const int k = tid;
        float a = bc0[k];
        float c = bi0[k];
        for (int f = 0; f < F_; ++f) {
            const float g = sU[f];
            a += g * Wc0u[f * K_ + k];
            c += g * Wi0u[f * K_ + k];
        }
        suC[b * K_ + k] = a;
        suI[b * K_ + k] = c;
    }
    float* og_u = out + B_;
    float* og_i = out + B_ + B_ * F_;
    float* op_i = out + B_ + 2 * B_ * F_;
    for (int f = tid; f < F_; f += 128) {
        og_u[b * F_ + f] = sU[f];
        og_i[b * F_ + f] = Gi[(size_t)it * F_ + f];
        op_i[b * F_ + f] = Pi[(size_t)it * F_ + f];
    }
}

// ---------------------------------------------------------------------------
// Kernel 2: component-level attention. One block per (b, p) pair.
// t = f16(f[49x128]) @ f16(Wc0i[128x64]) via v_wmma_f32_16x16x32_f16 (f32 acc).
// LDS staging is fragment-friendly: A rows f16 (zero-padded to 64 rows),
// B transposed f16, both with 136-half row stride so ds_load_b128 fragment
// loads hit all 64 banks. Epilogue: +proj, relu, dot Wc1, softmax(L), beta@f.
// ---------------------------------------------------------------------------
__global__ __launch_bounds__(128) void comp_attn(
    const int* __restrict__ pos_items,
    const float* __restrict__ Fi,
    const float* __restrict__ Wc0i,
    const float* __restrict__ Wc1, const float* __restrict__ bc1,
    const float* __restrict__ suC,
    float* __restrict__ allx)
{
    const int bp  = blockIdx.x;          // b*P + p
    const int b   = bp / P_;
    const int tid = threadIdx.x;
    const int lane = tid & 31;
    const int wave = tid >> 5;           // M-tile index (rows 16w..16w+15)

    alignas(16) __shared__ float    sF[L_ * C_];      // fp32 tile for beta@f
    alignas(16) __shared__ _Float16 sFh[MPAD * AST];  // f16 A tile, padded
    alignas(16) __shared__ _Float16 sWt[K_ * AST];    // f16 W^T [n][k]
    __shared__ float sL[64];

    // ---- stage f tile: coalesced float4 global loads -> fp32 + f16 LDS ----
    const int it = pos_items[bp];
    const float4* fsrc4 = (const float4*)(Fi + (size_t)it * (L_ * C_));
    for (int i = tid; i < (L_ * C_) / 4; i += 128) {   // 1568 float4
        const float4 v = fsrc4[i];
        ((float4*)sF)[i] = v;
        v4h h;
        h[0] = (_Float16)v.x; h[1] = (_Float16)v.y;
        h[2] = (_Float16)v.z; h[3] = (_Float16)v.w;
        *reinterpret_cast<v4h*>(&sFh[(i >> 5) * AST + (i & 31) * 4]) = h;
    }
    // zero pad rows 49..63 of the f16 A tile
    for (int i = tid; i < (MPAD - L_) * AST; i += 128)
        sFh[L_ * AST + i] = (_Float16)0.0f;
    // stage W^T: thread tid walks column k=tid over n; contiguous LDS writes
    for (int n = tid >> 7; n < K_; ++n)   // tid>>7 == 0: n = 0..63
        sWt[n * AST + tid] = (_Float16)Wc0i[tid * K_ + n];
    __syncthreads();

    // Fragment lane geometry (contiguous k order used here; A and B match):
    //  A 16x32: lane L, elem e -> M = L%16, K = e + 16*(L>=16)
    //  B 32x16: lane L, elem e -> N = L%16, K = e + 16*(L>=16)
    //  C/D 16x16 f32: elem r  -> M = r + 8*(L>=16), N = L%16
    const int mrow  = 16 * wave + (lane & 15);
    const int aKoff = (lane & 16);
    const int bKoff = (lane & 16);
    const int ncol  = lane & 15;

    v8f acc[4] = {};
#pragma unroll
    for (int ks = 0; ks < 4; ++ks) {        // K = 128 in 4 steps of 32
        const v8h* ap = reinterpret_cast<const v8h*>(&sFh[mrow * AST + 32 * ks + aKoff]);
        const v8h alo = ap[0];              // K base+0..7   (elems 0..7)
        const v8h ahi = ap[1];              // K base+8..15  (elems 8..15)
        const v16h af = __builtin_shufflevector(alo, ahi,
            0, 1, 2, 3, 4, 5, 6, 7, 8, 9, 10, 11, 12, 13, 14, 15);
#pragma unroll
        for (int n = 0; n < 4; ++n) {       // N = 64 in 4 tiles of 16
            const v8h* bp_ = reinterpret_cast<const v8h*>(
                &sWt[(16 * n + ncol) * AST + 32 * ks + bKoff]);
            const v8h blo = bp_[0];
            const v8h bhi = bp_[1];
            const v16h bf = __builtin_shufflevector(blo, bhi,
                0, 1, 2, 3, 4, 5, 6, 7, 8, 9, 10, 11, 12, 13, 14, 15);
            acc[n] = __builtin_amdgcn_wmma_f32_16x16x32_f16(
                false, af, false, bf, (short)0, acc[n], false, false);
        }
    }

    // Epilogue: relu(t + suC[b]) . Wc1  -> per-row logits
    float rs[8] = {0.f, 0.f, 0.f, 0.f, 0.f, 0.f, 0.f, 0.f};
#pragma unroll
    for (int n = 0; n < 4; ++n) {
        const int col = 16 * n + ncol;
        const float su = suC[b * K_ + col];  // includes bc0
        const float w1 = Wc1[col];
#pragma unroll
        for (int r = 0; r < 8; ++r) {
            float v = acc[n][r] + su;
            v = fmaxf(v, 0.f);
            rs[r] += v * w1;
        }
    }
#pragma unroll
    for (int off = 1; off <= 8; off <<= 1)
#pragma unroll
        for (int r = 0; r < 8; ++r) rs[r] += __shfl_xor(rs[r], off, 32);
    if ((lane & 15) == 0) {
        const int base = 16 * wave + ((lane >> 4) & 1) * 8;
        const float b1 = bc1[0];
#pragma unroll
        for (int r = 0; r < 8; ++r) sL[base + r] = rs[r] + b1;
    }
    __syncthreads();

    // softmax over L=49 components (wave 0)
    if (tid < 32) {
        float v1 = (tid      < L_) ? sL[tid]      : -1e30f;
        float v2 = (tid + 32 < L_) ? sL[tid + 32] : -1e30f;
        float m = fmaxf(v1, v2);
        for (int off = 16; off > 0; off >>= 1) m = fmaxf(m, __shfl_xor(m, off, 32));
        float e1 = (tid      < L_) ? __expf(v1 - m) : 0.f;
        float e2 = (tid + 32 < L_) ? __expf(v2 - m) : 0.f;
        float s = e1 + e2;
        for (int off = 16; off > 0; off >>= 1) s += __shfl_xor(s, off, 32);
        if (tid      < L_) sL[tid]      = e1 / s;
        if (tid + 32 < L_) sL[tid + 32] = e2 / s;
    }
    __syncthreads();

    // all_x[c] = sum_l beta[l] * f[l][c]   (fp32 from the LDS tile)
    {
        const int c = tid;
        float a = 0.f;
        for (int l = 0; l < L_; ++l) a += sL[l] * sF[l * C_ + c];
        allx[(size_t)bp * C_ + c] = a;
    }
}

// ---------------------------------------------------------------------------
// Kernel 3: item-level attention + final score. One block per user b.
// ---------------------------------------------------------------------------
__global__ __launch_bounds__(128) void item_attn(
    const int* __restrict__ user, const int* __restrict__ item,
    const int* __restrict__ pos_items, const unsigned char* __restrict__ pos_mask,
    const float* __restrict__ Gu, const float* __restrict__ Gi,
    const float* __restrict__ Pi,
    const float* __restrict__ Wi0iv, const float* __restrict__ Wi0ip,
    const float* __restrict__ Wi0ix,
    const float* __restrict__ Wi1, const float* __restrict__ bi1,
    const float* __restrict__ suI, const float* __restrict__ allx,
    float* __restrict__ out_xui)
{
    const int b = blockIdx.x;
    const int tid = threadIdx.x;
    __shared__ float sPi[P_ * F_];   // 40 KB: pi_p rows, reused for alpha @ pi_p
    __shared__ float sG[F_];
    __shared__ float sX[C_];
    __shared__ float sGam[F_];
    __shared__ float sGi[F_];
    __shared__ float sAl[P_ + 2];
    __shared__ float sRed[4];

    const int u  = user[b];
    const int it = item[b];
    for (int f = tid; f < F_; f += 128) {
        sGam[f] = Gu[(size_t)u * F_ + f];
        sGi[f]  = Gi[(size_t)it * F_ + f];
    }

    for (int p = 0; p < P_; ++p) {
        const int ip = pos_items[b * P_ + p];
        for (int f = tid; f < F_; f += 128) {
            sG[f]            = Gi[(size_t)ip * F_ + f];
            sPi[p * F_ + f]  = Pi[(size_t)ip * F_ + f];
        }
        if (tid < C_) sX[tid] = allx[((size_t)b * P_ + p) * C_ + tid];
        __syncthreads();
        if (tid < K_) {
            const int k = tid;
            float a = suI[b * K_ + k];       // includes bi0
            for (int f = 0; f < F_; ++f)
                a += sG[f] * Wi0iv[f * K_ + k] + sPi[p * F_ + f] * Wi0ip[f * K_ + k];
            for (int c = 0; c < C_; ++c)
                a += sX[c] * Wi0ix[c * K_ + k];
            a = fmaxf(a, 0.f);
            float contrib = a * Wi1[k];
            for (int off = 16; off > 0; off >>= 1)
                contrib += __shfl_xor(contrib, off, 32);
            if ((tid & 31) == 0) sRed[tid >> 5] = contrib;
        }
        __syncthreads();
        if (tid == 0) sAl[p] = sRed[0] + sRed[1] + bi1[0];
        __syncthreads();
    }

    // masked softmax over P=50 positives (wave 0)
    if (tid < 32) {
        float v1 = -1e30f, v2 = -1e30f;
        if (tid < P_)      v1 = pos_mask[b * P_ + tid]      ? sAl[tid]      : -1e9f;
        if (tid + 32 < P_) v2 = pos_mask[b * P_ + tid + 32] ? sAl[tid + 32] : -1e9f;
        float m = fmaxf(v1, v2);
        for (int off = 16; off > 0; off >>= 1) m = fmaxf(m, __shfl_xor(m, off, 32));
        float e1 = (tid      < P_) ? __expf(v1 - m) : 0.f;
        float e2 = (tid + 32 < P_) ? __expf(v2 - m) : 0.f;
        float s = e1 + e2;
        for (int off = 16; off > 0; off >>= 1) s += __shfl_xor(s, off, 32);
        if (tid      < P_) sAl[tid]      = e1 / s;
        if (tid + 32 < P_) sAl[tid + 32] = e2 / s;
    }
    __syncthreads();

    // xui = sum_f (gamma_u[f] + sum_p alpha[p]*pi_p[f]) * gamma_i[f]
    float part = 0.f;
    for (int f = tid; f < F_; f += 128) {
        float acc = 0.f;
        for (int p = 0; p < P_; ++p) acc += sAl[p] * sPi[p * F_ + f];
        part += (sGam[f] + acc) * sGi[f];
    }
    for (int off = 16; off > 0; off >>= 1) part += __shfl_xor(part, off, 32);
    if ((tid & 31) == 0) sRed[tid >> 5] = part;
    __syncthreads();
    if (tid == 0) out_xui[b] = sRed[0] + sRed[1] + sRed[2] + sRed[3];
}

// ---------------------------------------------------------------------------
extern "C" void kernel_launch(void* const* d_in, const int* in_sizes, int n_in,
                              void* d_out, int out_size, void* d_ws, size_t ws_size,
                              hipStream_t stream) {
    (void)in_sizes; (void)n_in; (void)out_size; (void)ws_size;
    const int*   user      = (const int*)d_in[0];
    const int*   item      = (const int*)d_in[1];
    const int*   pos_items = (const int*)d_in[2];
    const unsigned char* pos_mask = (const unsigned char*)d_in[3];
    const float* Gu    = (const float*)d_in[4];
    const float* Gi    = (const float*)d_in[5];
    const float* Pi    = (const float*)d_in[6];
    const float* Fi    = (const float*)d_in[7];
    const float* Wc0u  = (const float*)d_in[8];
    const float* Wc0i  = (const float*)d_in[9];
    const float* bc0   = (const float*)d_in[10];
    const float* Wc1   = (const float*)d_in[11];
    const float* bc1   = (const float*)d_in[12];
    const float* Wi0u  = (const float*)d_in[13];
    const float* Wi0iv = (const float*)d_in[14];
    const float* Wi0ip = (const float*)d_in[15];
    const float* Wi0ix = (const float*)d_in[16];
    const float* bi0   = (const float*)d_in[17];
    const float* Wi1   = (const float*)d_in[18];
    const float* bi1   = (const float*)d_in[19];

    float* out = (float*)d_out;
    float* ws  = (float*)d_ws;
    float* suC  = ws;                     // 128*64 floats
    float* suI  = ws + B_ * K_;           // 128*64 floats
    float* allx = ws + 2 * B_ * K_;       // 128*50*128 floats (~3.3 MB)

    user_proj<<<B_, 128, 0, stream>>>(user, item, Gu, Gi, Pi,
                                      Wc0u, bc0, Wi0u, bi0, suC, suI, out);
    comp_attn<<<B_ * P_, 128, 0, stream>>>(pos_items, Fi, Wc0i, Wc1, bc1,
                                           suC, allx);
    item_attn<<<B_, 128, 0, stream>>>(user, item, pos_items, pos_mask,
                                      Gu, Gi, Pi, Wi0iv, Wi0ip, Wi0ix,
                                      Wi1, bi1, suI, allx, out);
}
